// MultiHeadSelfAttn_67920612819221
// MI455X (gfx1250) — matrix operand order repaired and checked by reading, not compiled, not run on hardware
//
#include <hip/hip_runtime.h>
#include <hip/hip_bf16.h>
#include <math.h>

typedef __attribute__((ext_vector_type(16))) _Float16 v16h;
typedef __attribute__((ext_vector_type(8)))  _Float16 v8h;
typedef __attribute__((ext_vector_type(4)))  _Float16 v4h;
typedef __attribute__((ext_vector_type(8)))  float    v8f;

#define D_MODEL 1024
#define N_HEADS 16
#define D_HEAD  64
#define BATCH   2
#define SEQ     2048
#define M_TOTAL (BATCH * SEQ)

// LDS tile geometry (halves). Row stride 48 keeps 16B/32B fragment alignment
// and staggers banks.
#define LDS_STRIDE 48
#define SLAB_A (64 * LDS_STRIDE)    // 64-row A tile
#define SLAB_B (256 * LDS_STRIDE)   // 256-col B tile (stored [n][k])
#define SLAB   (SLAB_A + SLAB_B)

// ---------------------------------------------------------------------------
// WMMA helpers (wave32; contiguous k = 16h + s, same order in A and B)
// ---------------------------------------------------------------------------
__device__ __forceinline__ v16h make_v16(v8h lo, v8h hi) {
    return __builtin_shufflevector(lo, hi, 0, 1, 2, 3, 4, 5, 6, 7,
                                           8, 9, 10, 11, 12, 13, 14, 15);
}

__device__ __forceinline__ v8f wmma_f16(v16h a, v16h b, v8f c) {
    return __builtin_amdgcn_wmma_f32_16x16x32_f16(
        false, a, false, b, (short)0, c, false, false);
}

// A fragment 16x32 f16: lane holds row m = lane&15.
// lanes 0-15: k = 0..15; lanes 16-31: k = 16..31.
__device__ __forceinline__ v16h load_afrag(const _Float16* rowPtr, int lane) {
    const int kBase = (lane < 16) ? 0 : 16;
    v8h lo = *(const v8h*)(rowPtr + kBase);
    v8h hi = *(const v8h*)(rowPtr + kBase + 8);
    return make_v16(lo, hi);
}

// B fragment 32x16 f16: lane holds column n = lane&15; k contiguous per lane.
__device__ __forceinline__ v16h load_bfrag(const _Float16* colPtr, int lane) {
    const int kBase = (lane < 16) ? 0 : 16;
    return *(const v16h*)(colPtr + kBase);
}

// ---------------------------------------------------------------------------
// CDNA5 async memory->LDS copy (ASYNCcnt-tracked), 16 bytes per lane.
// VDST holds the LDS destination byte address; generic-pointer low 32 bits are the
// wave-local LDS byte offset.
// ---------------------------------------------------------------------------
__device__ __forceinline__ unsigned lds_off(const void* p) {
    return (unsigned)(size_t)p;
}

__device__ __forceinline__ void async_b128(unsigned ldsByteOff,
                                           const _Float16* gptr) {
    asm volatile("global_load_async_to_lds_b128 %0, %1, off"
                 :: "v"(ldsByteOff), "v"(gptr)
                 : "memory");
}

// ---------------------------------------------------------------------------
// Double-buffered async tiled GEMM core:
//   C[64 x 256] = A[64 x 1024] * W^T      (A, W both f16; W is [N,K] row-major)
// Block = 256 threads = 8 waves; wave (wm, wn) owns 32x64 (2x4 WMMA tiles).
// Per k-step (K=32): 5 async b128 copies/thread for next buffer,
// s_wait_asynccnt 5, then 8 WMMAs from current buffer.
// ---------------------------------------------------------------------------
__device__ __forceinline__ void stage_async(const _Float16* __restrict__ A,
                                            const _Float16* __restrict__ W,
                                            _Float16* buf, int m0, int n0,
                                            int kt, int tid) {
    // A tile: 64 rows x 32 halves = 256 x 16B units, 1 unit per thread
    {
        const int row = tid >> 2;
        const int kk  = (tid & 3) * 8;
        async_b128(lds_off(buf + row * LDS_STRIDE + kk),
                   A + (size_t)(m0 + row) * D_MODEL + kt + kk);
    }
    // B tile: 256 rows(n) x 32 halves = 1024 units, 4 per thread
    _Float16* bufB = buf + SLAB_A;
#pragma unroll
    for (int it = 0; it < 4; ++it) {
        const int u   = it * 256 + tid;
        const int n   = u >> 2;
        const int kk  = (u & 3) * 8;
        async_b128(lds_off(bufB + n * LDS_STRIDE + kk),
                   W + (size_t)(n0 + n) * D_MODEL + kt + kk);
    }
}

__device__ __forceinline__ void gemm_block(const _Float16* __restrict__ A,
                                           const _Float16* __restrict__ W,
                                           _Float16* lds,  // 2 * SLAB halves
                                           int m0, int n0, v8f acc[2][4]) {
    const int tid   = threadIdx.x;
    const int lane  = tid & 31;
    const int wave  = tid >> 5;
    const int waveM = wave & 1;   // 2 x 32 rows
    const int waveN = wave >> 1;  // 4 x 64 cols

    stage_async(A, W, lds, m0, n0, 0, tid);

    int cur = 0;
    for (int kt = 0; kt < D_MODEL; kt += 32) {
        const int nxt = kt + 32;
        if (nxt < D_MODEL) {
            stage_async(A, W, lds + (cur ^ 1) * SLAB, m0, n0, nxt, tid);
            asm volatile("s_wait_asynccnt 0x5" ::: "memory");
        } else {
            asm volatile("s_wait_asynccnt 0x0" ::: "memory");
        }
        __syncthreads();

        _Float16* ldsA = lds + cur * SLAB;
        _Float16* ldsB = ldsA + SLAB_A;

        v16h aF[2], bF[4];
#pragma unroll
        for (int i = 0; i < 2; ++i) {
            const int m = waveM * 32 + i * 16 + (lane & 15);
            aF[i] = load_afrag(ldsA + m * LDS_STRIDE, lane);
        }
#pragma unroll
        for (int j = 0; j < 4; ++j) {
            const int n = waveN * 64 + j * 16 + (lane & 15);
            bF[j] = load_bfrag(ldsB + n * LDS_STRIDE, lane);
        }
#pragma unroll
        for (int i = 0; i < 2; ++i)
#pragma unroll
            for (int j = 0; j < 4; ++j)
                acc[i][j] = wmma_f16(aF[i], bF[j], acc[i][j]);
        __syncthreads();
        cur ^= 1;
    }
}

// ---------------------------------------------------------------------------
// Kernel 0: f32 -> f16 conversion (H and the four weight matrices)
// ---------------------------------------------------------------------------
__global__ __launch_bounds__(256) void cvt_f16_kernel(
    const float* __restrict__ src, _Float16* __restrict__ dst, int n4) {
    const int i = blockIdx.x * 256 + threadIdx.x;
    if (i < n4) {
        const float4 v = ((const float4*)src)[i];
        v4h o;
        o.x = (_Float16)v.x; o.y = (_Float16)v.y;
        o.z = (_Float16)v.z; o.w = (_Float16)v.w;
        ((v4h*)dst)[i] = o;
    }
}

// ---------------------------------------------------------------------------
// Kernel 1: Q/K/V projections (blockIdx.z selects projection).
//   Q -> [B, nh, T, dh] f16, pre-scaled by 1/sqrt(dh)
//   K -> [B, nh, T, dh] f16
//   V -> [B, nh, dh, T] f16 (transposed: contiguous P*V B-fragments)
// ---------------------------------------------------------------------------
__global__ __launch_bounds__(256) void qkv_proj_kernel(
    const _Float16* __restrict__ Hh,
    const _Float16* __restrict__ Wqh, const _Float16* __restrict__ Wkh,
    const _Float16* __restrict__ Wvh,
    const float* __restrict__ bq, const float* __restrict__ bk,
    const float* __restrict__ bv,
    _Float16* __restrict__ Q, _Float16* __restrict__ K,
    _Float16* __restrict__ Vt) {
    __shared__ _Float16 lds[2 * SLAB];

    const int n0 = blockIdx.x * 256;
    const int m0 = blockIdx.y * 64;
    const int z  = blockIdx.z;
    const _Float16* W  = (z == 0) ? Wqh : (z == 1) ? Wkh : Wvh;
    const float* bias  = (z == 0) ? bq  : (z == 1) ? bk  : bv;

    v8f acc[2][4] = {};
    gemm_block(Hh, W, lds, m0, n0, acc);

    const int tid = threadIdx.x, lane = tid & 31, wave = tid >> 5;
    const int waveM = wave & 1, waveN = wave >> 1;
    const int rowOff = (lane < 16) ? 0 : 8;

#pragma unroll
    for (int j = 0; j < 4; ++j) {
        const int n = n0 + waveN * 64 + j * 16 + (lane & 15);
        const float bn = bias[n];
        const int head = n >> 6, d = n & 63;
#pragma unroll
        for (int i = 0; i < 2; ++i) {
#pragma unroll
            for (int r = 0; r < 8; ++r) {
                const int m = m0 + waveM * 32 + i * 16 + r + rowOff;
                const int b = m >> 11;
                const int t = m & (SEQ - 1);
                const float v = acc[i][j][r] + bn;
                const size_t bh = (size_t)(b * N_HEADS + head);
                if (z == 0)
                    Q[(bh * SEQ + t) * D_HEAD + d] = (_Float16)(v * 0.125f);
                else if (z == 1)
                    K[(bh * SEQ + t) * D_HEAD + d] = (_Float16)v;
                else
                    Vt[(bh * D_HEAD + d) * SEQ + t] = (_Float16)v;
            }
        }
    }
}

// ---------------------------------------------------------------------------
// Kernel 2: causal flash attention. 4 independent waves per block, each owns
// one 16-row Q tile and a private LDS slice for the P re-layout.
// ---------------------------------------------------------------------------
__global__ __launch_bounds__(128) void flash_attn_kernel(
    const _Float16* __restrict__ Q, const _Float16* __restrict__ K,
    const _Float16* __restrict__ Vt, const unsigned char* __restrict__ kpm,
    _Float16* __restrict__ Ocat) {
    __shared__ _Float16 ldsP[4][16 * 32];

    const int lane = threadIdx.x & 31;
    const int wv   = threadIdx.x >> 5;
    const int qt = blockIdx.x * 4 + wv;
    const int bh = blockIdx.y;
    const int b = bh >> 4;
    const int h = bh & 15;

    const _Float16* Qh  = Q  + (size_t)bh * SEQ * D_HEAD;
    const _Float16* Kh  = K  + (size_t)bh * SEQ * D_HEAD;
    const _Float16* Vth = Vt + (size_t)bh * D_HEAD * SEQ;
    _Float16* Pw = ldsP[wv];

    const int mA = lane & 15;
    v16h aQ[2];
#pragma unroll
    for (int s = 0; s < 2; ++s)
        aQ[s] = load_afrag(Qh + (size_t)(qt * 16 + mA) * D_HEAD + s * 32, lane);

    v8f oAcc[4] = {};
    float mrow[8], lrow[8];
#pragma unroll
    for (int r = 0; r < 8; ++r) { mrow[r] = -3.0e30f; lrow[r] = 0.0f; }

    const int rowOff = (lane < 16) ? 0 : 8;
    const int nCol = lane & 15;
    const int ntiles = (qt * 16 + 15) / 32 + 1;

    for (int kt = 0; kt < ntiles; ++kt) {
        const int kv0 = kt * 32;

        v8f sAcc[2] = {};
#pragma unroll
        for (int s = 0; s < 2; ++s) {
#pragma unroll
            for (int ns = 0; ns < 2; ++ns) {
                const int kv = kv0 + ns * 16 + nCol;
                v16h bK = load_bfrag(Kh + (size_t)kv * D_HEAD + s * 32, lane);
                sAcc[ns] = wmma_f16(aQ[s], bK, sAcc[ns]);
            }
        }

        float pv[2][8];
#pragma unroll
        for (int ns = 0; ns < 2; ++ns) {
            const int kv = kv0 + ns * 16 + nCol;
            const bool pad = kpm[(size_t)b * SEQ + kv] != 0;
#pragma unroll
            for (int r = 0; r < 8; ++r) {
                const int q = qt * 16 + r + rowOff;
                float sv = sAcc[ns][r];
                if (kv > q) sv -= 1.0e9f;   // soft causal mask, as reference
                if (pad)    sv = -3.0e38f;  // key-padding -> -inf
                sAcc[ns][r] = sv;
            }
        }

#pragma unroll
        for (int r = 0; r < 8; ++r) {
            float v = fmaxf(sAcc[0][r], sAcc[1][r]);
#pragma unroll
            for (int off = 1; off < 16; off <<= 1)
                v = fmaxf(v, __shfl_xor(v, off, 32));
            const float mNew = fmaxf(mrow[r], v);
            const float corr = __expf(mrow[r] - mNew);
            mrow[r] = mNew;
            const float p0 = __expf(sAcc[0][r] - mNew);
            const float p1 = __expf(sAcc[1][r] - mNew);
            pv[0][r] = p0; pv[1][r] = p1;
            float rs = p0 + p1;
#pragma unroll
            for (int off = 1; off < 16; off <<= 1)
                rs += __shfl_xor(rs, off, 32);
            lrow[r] = lrow[r] * corr + rs;
#pragma unroll
            for (int ds = 0; ds < 4; ++ds) oAcc[ds][r] *= corr;
        }

#pragma unroll
        for (int r = 0; r < 8; ++r) {
            const int m = r + rowOff;
            Pw[m * 32 + nCol]      = (_Float16)pv[0][r];
            Pw[m * 32 + 16 + nCol] = (_Float16)pv[1][r];
        }
        asm volatile("s_wait_dscnt 0x0" ::: "memory");
        __builtin_amdgcn_wave_barrier();

        const v16h aP = load_afrag(Pw + mA * 32, lane);
#pragma unroll
        for (int ds = 0; ds < 4; ++ds) {
            const int d = ds * 16 + nCol;
            v16h bV = load_bfrag(Vth + (size_t)d * SEQ + kv0, lane);
            oAcc[ds] = wmma_f16(aP, bV, oAcc[ds]);
        }
        __builtin_amdgcn_wave_barrier();
    }

#pragma unroll
    for (int r = 0; r < 8; ++r) {
        const int t = qt * 16 + r + rowOff;
        const float inv = 1.0f / lrow[r];
        const size_t base = ((size_t)b * SEQ + t) * D_MODEL + h * D_HEAD;
#pragma unroll
        for (int ds = 0; ds < 4; ++ds)
            Ocat[base + ds * 16 + nCol] = (_Float16)(oAcc[ds][r] * inv);
    }
}

// ---------------------------------------------------------------------------
// Kernel 3: output projection Y = Ocat @ Wo^T + bo (f32 output)
// ---------------------------------------------------------------------------
__global__ __launch_bounds__(256) void out_proj_kernel(
    const _Float16* __restrict__ Ocat, const _Float16* __restrict__ Woh,
    const float* __restrict__ bo, float* __restrict__ Y) {
    __shared__ _Float16 lds[2 * SLAB];

    const int n0 = blockIdx.x * 256;
    const int m0 = blockIdx.y * 64;

    v8f acc[2][4] = {};
    gemm_block(Ocat, Woh, lds, m0, n0, acc);

    const int tid = threadIdx.x, lane = tid & 31, wave = tid >> 5;
    const int waveM = wave & 1, waveN = wave >> 1;
    const int rowOff = (lane < 16) ? 0 : 8;

#pragma unroll
    for (int j = 0; j < 4; ++j) {
        const int n = n0 + waveN * 64 + j * 16 + (lane & 15);
        const float bn = bo[n];
#pragma unroll
        for (int i = 0; i < 2; ++i) {
#pragma unroll
            for (int r = 0; r < 8; ++r) {
                const int m = m0 + waveM * 32 + i * 16 + r + rowOff;
                Y[(size_t)m * D_MODEL + n] = acc[i][j][r] + bn;
            }
        }
    }
}

// ---------------------------------------------------------------------------
// Launch
// ---------------------------------------------------------------------------
extern "C" void kernel_launch(void* const* d_in, const int* in_sizes, int n_in,
                              void* d_out, int out_size, void* d_ws,
                              size_t ws_size, hipStream_t stream) {
    (void)in_sizes; (void)n_in; (void)out_size; (void)ws_size;

    const float* H          = (const float*)d_in[0];
    const unsigned char* km = (const unsigned char*)d_in[1];
    const float* Wq = (const float*)d_in[2];
    const float* bq = (const float*)d_in[3];
    const float* Wk = (const float*)d_in[4];
    const float* bk = (const float*)d_in[5];
    const float* Wv = (const float*)d_in[6];
    const float* bv = (const float*)d_in[7];
    const float* Wo = (const float*)d_in[8];
    const float* bo = (const float*)d_in[9];

    _Float16* ws = (_Float16*)d_ws;
    const size_t per  = (size_t)BATCH * N_HEADS * SEQ * D_HEAD;  // 4,194,304
    const size_t wsz  = (size_t)D_MODEL * D_MODEL;               // 1,048,576
    _Float16* Q    = ws;
    _Float16* K    = ws + per;
    _Float16* Vt   = ws + 2 * per;
    _Float16* Ocat = ws + 3 * per;
    _Float16* Hh   = ws + 4 * per;
    _Float16* Wqh  = Hh + per;           // H is also `per` elements (4M)
    _Float16* Wkh  = Wqh + wsz;
    _Float16* Wvh  = Wkh + wsz;
    _Float16* Woh  = Wvh + wsz;

    // Kernel 0: one-shot f32 -> f16 conversions
    cvt_f16_kernel<<<dim3((int)(per / 4 / 256)), 256, 0, stream>>>(H,  Hh,  (int)(per / 4));
    cvt_f16_kernel<<<dim3((int)(wsz / 4 / 256)), 256, 0, stream>>>(Wq, Wqh, (int)(wsz / 4));
    cvt_f16_kernel<<<dim3((int)(wsz / 4 / 256)), 256, 0, stream>>>(Wk, Wkh, (int)(wsz / 4));
    cvt_f16_kernel<<<dim3((int)(wsz / 4 / 256)), 256, 0, stream>>>(Wv, Wvh, (int)(wsz / 4));
    cvt_f16_kernel<<<dim3((int)(wsz / 4 / 256)), 256, 0, stream>>>(Wo, Woh, (int)(wsz / 4));

    dim3 g1(D_MODEL / 256, M_TOTAL / 64, 3);
    qkv_proj_kernel<<<g1, 256, 0, stream>>>(Hh, Wqh, Wkh, Wvh, bq, bk, bv,
                                            Q, K, Vt);

    dim3 g2(SEQ / 64, BATCH * N_HEADS);
    flash_attn_kernel<<<g2, 128, 0, stream>>>(Q, K, Vt, km, Ocat);

    dim3 g3(D_MODEL / 256, M_TOTAL / 64);
    out_proj_kernel<<<g3, 256, 0, stream>>>(Ocat, Woh, bo, (float*)d_out);
}
